// MPMC_net_54803782696982
// MI455X (gfx1250) — matrix operand order repaired and checked by reading, not compiled, not run on hardware
//
#include <hip/hip_runtime.h>
#include <hip/hip_bf16.h>
#include <stdint.h>

#define NPTS 4096
#define HID  128

typedef __attribute__((ext_vector_type(16))) __bf16 v16bf;
typedef __attribute__((ext_vector_type(8)))  __bf16 v8bf;
typedef __attribute__((ext_vector_type(8)))  float  v8f;
typedef uint32_t u32x4 __attribute__((ext_vector_type(4)));
typedef uint32_t u32x8 __attribute__((ext_vector_type(8)));

static __device__ inline v16bf cat16(v8bf lo, v8bf hi) {
  return __builtin_shufflevector(lo, hi, 0,1,2,3,4,5,6,7,8,9,10,11,12,13,14,15);
}

// A fragment (16x32 bf16) from LDS row-major [16 x ldK]:
// lanes 0-15: M=lane, K = kt*32 + {0..7, 16..23}
// lanes 16-31: M=lane-16, K = kt*32 + {8..15, 24..31}
static __device__ inline v16bf load_a_frag(const __bf16* A, int ldK, int kt, int lane) {
  int row  = lane & 15;
  int koff = kt * 32 + ((lane >> 4) << 3);
  const __bf16* p = A + row * ldK + koff;
  v8bf lo = *(const v8bf*)(p);
  v8bf hi = *(const v8bf*)(p + 16);
  return cat16(lo, hi);
}

// B fragment: pre-swizzled so each lane's 16 bf16 are contiguous (32 bytes).
static __device__ inline v16bf load_b_frag(const __bf16* __restrict__ frag, int lane) {
  const v8bf* q = (const v8bf*)(frag + lane * 16);
  return cat16(q[0], q[1]);
}

// Swizzle fp32 weights [K x 128] row-major into bf16 WMMA B-fragments.
__global__ void swizzle_B(const float* __restrict__ W, __bf16* __restrict__ out, int K) {
  int t = blockIdx.x * blockDim.x + threadIdx.x;
  int nfrag = (K / 32) * 8;
  if (t >= nfrag * 32) return;
  int frag = t >> 5, lane = t & 31;
  int kt = frag >> 3, nt = frag & 7;
  int kbase = kt * 32 + ((lane >> 4) << 3);
  int col   = nt * 16 + (lane & 15);
  __bf16* o = out + (size_t)frag * 512 + lane * 16;
#pragma unroll
  for (int j = 0; j < 16; ++j) o[j] = (__bf16)W[(size_t)(kbase + j + ((j >> 3) << 3)) * HID + col];
}

__global__ void encode_kernel(const float* __restrict__ P, const float* __restrict__ W,
                              const float* __restrict__ b, float* __restrict__ h,
                              __bf16* __restrict__ hb) {
  int t = blockIdx.x * 256 + threadIdx.x;      // NPTS*HID threads
  int n = t >> 7, c = t & (HID - 1);
  float acc = b[c];
#pragma unroll
  for (int d = 0; d < 6; ++d) acc += P[n * 6 + d] * W[d * HID + c];
  h[t]  = acc;
  hb[t] = (__bf16)acc;
}

// Edge MLP: m = relu(relu([x_dst | x_src] @ W1 + b1) @ W2 + b2); agg[dst] += m
__global__ void __launch_bounds__(256)
msg_kernel(const __bf16* __restrict__ hb, const int* __restrict__ src,
           const int* __restrict__ dst,
           const __bf16* __restrict__ W1, const float* __restrict__ b1,
           const __bf16* __restrict__ W2, const float* __restrict__ b2,
           float* __restrict__ agg, int E) {
  __shared__ __align__(32) __bf16 Arows[16 * 256];
  __shared__ __align__(32) __bf16 M1[16 * HID];
  __shared__ int dIdx[16];
  const int tid = threadIdx.x;
  const int lane = tid & 31, wave = tid >> 5;
  const int e0 = blockIdx.x * 16;

  { // stage gathered edge features via async DMA straight into LDS (ASYNCcnt path)
    int row = tid >> 4, seg = tid & 15;
    int e = e0 + row;
    uint32_t ldsa = (uint32_t)(uintptr_t)(&Arows[row * 256 + seg * 16]);
    if (e < E) {
      int s = src[e], d = dst[e];
      const __bf16* sp = (seg < 8) ? (hb + (size_t)d * HID + seg * 16)
                                   : (hb + (size_t)s * HID + (seg - 8) * 16);
      uint64_t ga = (uint64_t)(uintptr_t)sp;
      asm volatile("global_load_async_to_lds_b128 %0, %1, off\n\t"
                   "global_load_async_to_lds_b128 %0, %1, off offset:16"
                   :: "v"(ldsa), "v"(ga) : "memory");
      if (seg == 0) dIdx[row] = d;
    } else {
      __bf16* dp = &Arows[row * 256 + seg * 16];
#pragma unroll
      for (int j = 0; j < 16; ++j) dp[j] = (__bf16)0.0f;
      if (seg == 0) dIdx[row] = -1;
    }
  }
  asm volatile("s_wait_asynccnt 0" ::: "memory");
  __syncthreads();

  const int nt = wave;                          // wave owns one 16-wide N tile
  v8f acc = {};
#pragma unroll
  for (int kt = 0; kt < 8; ++kt) {              // K = 256
    v16bf a = load_a_frag(Arows, 256, kt, lane);
    const __bf16* bptr = W1 + (size_t)(kt * 8 + nt) * 512;
    if (kt < 7) __builtin_prefetch(W1 + (size_t)((kt + 1) * 8 + nt) * 512, 0, 1);
    v16bf b = load_b_frag(bptr, lane);
    acc = __builtin_amdgcn_wmma_f32_16x16x32_bf16(false, a, false, b, (short)0, acc,
                                                  false, false);
  }
  { // bias + relu + bf16 -> LDS intermediate
    int mofs = (lane >> 4) << 3;
    int n = nt * 16 + (lane & 15);
    float bias = b1[n];
#pragma unroll
    for (int r = 0; r < 8; ++r) {
      float v = fmaxf(acc[r] + bias, 0.f);
      M1[(r + mofs) * HID + n] = (__bf16)v;
    }
  }
  __syncthreads();

  v8f acc2 = {};
#pragma unroll
  for (int kt = 0; kt < 4; ++kt) {              // K = 128
    v16bf a = load_a_frag(M1, HID, kt, lane);
    v16bf b = load_b_frag(W2 + (size_t)(kt * 8 + nt) * 512, lane);
    acc2 = __builtin_amdgcn_wmma_f32_16x16x32_bf16(false, a, false, b, (short)0, acc2,
                                                   false, false);
  }
  { // bias + relu + scatter-add (segment_sum)
    int mofs = (lane >> 4) << 3;
    int n = nt * 16 + (lane & 15);
    float bias = b2[n];
#pragma unroll
    for (int r = 0; r < 8; ++r) {
      float v = fmaxf(acc2[r] + bias, 0.f);
      int d = dIdx[r + mofs];
      if (d >= 0) atomicAdd(&agg[(size_t)d * HID + n], v);
    }
  }
}

// Node MLP: u = relu(relu([x | agg] @ W1 + b1) @ W2 + b2)
__global__ void __launch_bounds__(256)
upd_kernel(const __bf16* __restrict__ hb, const float* __restrict__ agg,
           const __bf16* __restrict__ W1, const float* __restrict__ b1,
           const __bf16* __restrict__ W2, const float* __restrict__ b2,
           float* __restrict__ u) {
  __shared__ __align__(32) __bf16 Arows[16 * 256];
  __shared__ __align__(32) __bf16 M1[16 * HID];
  const int tid = threadIdx.x;
  const int lane = tid & 31, wave = tid >> 5;
  const int n0 = blockIdx.x * 16;

  // TDM: stage h rows [n0..n0+15][0..127] (bf16, 256 B each) into Arows with a
  // 256 B pad after every 256 B stored, interleaving space for the agg half.
  // D# per CDNA5 ISA:
  //   tile_dim0=128, tile_dim1=16, data_size=2B, tensor_dim0_stride=128,
  //   pad_enable=1, pad_interval=5 (32 x 8B = 256 B), pad_amount=63 (64 DW = 256 B)
  {
    uint64_t ga   = (uint64_t)(uintptr_t)(hb + (size_t)n0 * HID);
    uint32_t lbase = (uint32_t)(uintptr_t)(&Arows[0]);
    u32x4 g0;
    g0.x = 1u;                                               // count=1, user D#
    g0.y = lbase;                                            // lds_addr
    g0.z = (uint32_t)ga;                                     // global_addr[31:0]
    g0.w = (uint32_t)((ga >> 32) & 0x01FFFFFFu) | (2u << 30); // addr[56:32], type=2
    u32x8 g1;
    g1.s0 = (1u << 16) | (1u << 20) | (5u << 22) | (63u << 25); // data_size=2B, pad
    g1.s1 = (uint32_t)HID << 16;   // tensor_dim0 = 128     (bits 79:48 lo half)
    g1.s2 = (uint32_t)NPTS << 16;  // tensor_dim1 = 4096    (bits 111:80 lo half)
    g1.s3 = (uint32_t)HID << 16;   // tile_dim0 = 128       (bits 127:112)
    g1.s4 = 16u;                   // tile_dim1 = 16        (bits 143:128)
    g1.s5 = (uint32_t)HID;         // tensor_dim0_stride    (bits 207:160 lo)
    g1.s6 = 0u;
    g1.s7 = 0u;
    if (wave == 0) {
      asm volatile("tensor_load_to_lds %0, %1" :: "s"(g0), "s"(g1) : "memory");
    }
  }
  { // agg half (needs f32->bf16 conversion, so VALU + ds stores)
    int row = tid >> 4, seg = tid & 15;
    if (seg >= 8) {
      int node = n0 + row;
      __bf16* dp = &Arows[row * 256 + seg * 16];
      const float* ap = agg + (size_t)node * HID + (seg - 8) * 16;
#pragma unroll
      for (int j = 0; j < 16; ++j) dp[j] = (__bf16)ap[j];
    }
  }
  __builtin_amdgcn_s_wait_tensorcnt(0);
  __syncthreads();

  const int nt = wave;
  v8f acc = {};
#pragma unroll
  for (int kt = 0; kt < 8; ++kt) {
    v16bf a = load_a_frag(Arows, 256, kt, lane);
    v16bf b = load_b_frag(W1 + (size_t)(kt * 8 + nt) * 512, lane);
    acc = __builtin_amdgcn_wmma_f32_16x16x32_bf16(false, a, false, b, (short)0, acc,
                                                  false, false);
  }
  {
    int mofs = (lane >> 4) << 3;
    int n = nt * 16 + (lane & 15);
    float bias = b1[n];
#pragma unroll
    for (int r = 0; r < 8; ++r) {
      float v = fmaxf(acc[r] + bias, 0.f);
      M1[(r + mofs) * HID + n] = (__bf16)v;
    }
  }
  __syncthreads();

  v8f acc2 = {};
#pragma unroll
  for (int kt = 0; kt < 4; ++kt) {
    v16bf a = load_a_frag(M1, HID, kt, lane);
    v16bf b = load_b_frag(W2 + (size_t)(kt * 8 + nt) * 512, lane);
    acc2 = __builtin_amdgcn_wmma_f32_16x16x32_bf16(false, a, false, b, (short)0, acc2,
                                                   false, false);
  }
  {
    int mofs = (lane >> 4) << 3;
    int n = nt * 16 + (lane & 15);
    float bias = b2[n];
#pragma unroll
    for (int r = 0; r < 8; ++r) {
      float v = fmaxf(acc2[r] + bias, 0.f);
      u[(size_t)(n0 + r + mofs) * HID + n] = v;
    }
  }
}

__global__ void norm_stats_kernel(const float* __restrict__ u, float* __restrict__ mean,
                                  float* __restrict__ rstd) {
  __shared__ float ss[256], sq[256];
  int c = blockIdx.x, tid = threadIdx.x;
  float s = 0.f, q = 0.f;
  for (int n = tid; n < NPTS; n += 256) {
    float v = u[(size_t)n * HID + c];
    s += v; q += v * v;
  }
  ss[tid] = s; sq[tid] = q;
  __syncthreads();
  for (int o = 128; o > 0; o >>= 1) {
    if (tid < o) { ss[tid] += ss[tid + o]; sq[tid] += sq[tid + o]; }
    __syncthreads();
  }
  if (tid == 0) {
    float m = ss[0] / (float)NPTS;
    float var = sq[0] / (float)NPTS - m * m;
    mean[c] = m;
    rstd[c] = rsqrtf(var + 1e-5f);
  }
}

__global__ void norm_apply_kernel(const float* __restrict__ u, const float* __restrict__ mean,
                                  const float* __restrict__ rstd, float* __restrict__ h,
                                  __bf16* __restrict__ hb) {
  int t = blockIdx.x * 256 + threadIdx.x;
  int c = t & (HID - 1);
  float v = (u[t] - mean[c]) * rstd[c] + h[t];
  h[t]  = v;
  hb[t] = (__bf16)v;
}

__global__ void decode_kernel(const float* __restrict__ h, const float* __restrict__ W,
                              const float* __restrict__ b, float* __restrict__ X) {
  int n = blockIdx.x * blockDim.x + threadIdx.x;
  float acc[6];
#pragma unroll
  for (int d = 0; d < 6; ++d) acc[d] = b[d];
  for (int c = 0; c < HID; ++c) {
    float hv = h[(size_t)n * HID + c];
#pragma unroll
    for (int d = 0; d < 6; ++d) acc[d] += hv * W[c * 6 + d];
  }
#pragma unroll
  for (int d = 0; d < 6; ++d) X[n * 6 + d] = 1.f / (1.f + __expf(-acc[d]));
}

// sum1 = sum_i prod_d (1 - x_id^2);  sum2 = sum_ij prod_d (1 - max(x_id, x_jd))
__global__ void __launch_bounds__(256)
disc_kernel(const float* __restrict__ X, float* __restrict__ sums) {
  __shared__ float tile[256 * 6];
  __shared__ float red[256];
  int tid = threadIdx.x;
  int i = blockIdx.x * 256 + tid;
  float xi[6];
#pragma unroll
  for (int d = 0; d < 6; ++d) xi[d] = X[i * 6 + d];
  float p1 = 1.f;
#pragma unroll
  for (int d = 0; d < 6; ++d) p1 *= (1.f - xi[d] * xi[d]);
  float s2 = 0.f;
  for (int jt = 0; jt < NPTS / 256; ++jt) {
    __syncthreads();
#pragma unroll
    for (int d = 0; d < 6; ++d) tile[tid * 6 + d] = X[(jt * 256 + tid) * 6 + d];
    __syncthreads();
    for (int j = 0; j < 256; ++j) {
      float p = 1.f;
#pragma unroll
      for (int d = 0; d < 6; ++d) {
        float m = fmaxf(xi[d], tile[j * 6 + d]);
        p *= (1.f - m);
      }
      s2 += p;
    }
  }
  red[tid] = p1;
  __syncthreads();
  for (int o = 128; o > 0; o >>= 1) {
    if (tid < o) red[tid] += red[tid + o];
    __syncthreads();
  }
  if (tid == 0) atomicAdd(&sums[0], red[0]);
  __syncthreads();
  red[tid] = s2;
  __syncthreads();
  for (int o = 128; o > 0; o >>= 1) {
    if (tid < o) red[tid] += red[tid + o];
    __syncthreads();
  }
  if (tid == 0) atomicAdd(&sums[1], red[0]);
}

__global__ void finalize_kernel(const float* __restrict__ sums, float* __restrict__ out) {
  float term1 = 1.0f / 729.0f;                                   // 3^-6
  float term2 = (2.0f / (float)NPTS) * (1.0f / 32.0f) * sums[0]; // 2^(1-6)=1/32
  float term3 = sums[1] / ((float)NPTS * (float)NPTS);
  float v = term1 - term2 + term3;
  out[0] = sqrtf(fmaxf(v, 1e-8f));
}

extern "C" void kernel_launch(void* const* d_in, const int* in_sizes, int n_in,
                              void* d_out, int out_size, void* d_ws, size_t ws_size,
                              hipStream_t stream) {
  const float* P    = (const float*)d_in[0];
  const int*   eidx = (const int*)d_in[1];
  const float* encW = (const float*)d_in[2];
  const float* encB = (const float*)d_in[3];
  const float* m1W  = (const float*)d_in[4];
  const float* m1B  = (const float*)d_in[5];
  const float* m2W  = (const float*)d_in[6];
  const float* m2B  = (const float*)d_in[7];
  const float* u1W  = (const float*)d_in[8];
  const float* u1B  = (const float*)d_in[9];
  const float* u2W  = (const float*)d_in[10];
  const float* u2B  = (const float*)d_in[11];
  const float* decW = (const float*)d_in[12];
  const float* decB = (const float*)d_in[13];
  const int E = in_sizes[1] / 2;
  const int* src = eidx;
  const int* dst = eidx + E;

  char* p = (char*)d_ws;
  auto take = [&](size_t bytes) {
    void* r = (void*)p;
    p += (bytes + 255) & ~(size_t)255;
    return r;
  };
  float*  h    = (float*)take((size_t)NPTS * HID * 4);
  float*  u    = (float*)take((size_t)NPTS * HID * 4);
  float*  agg  = (float*)take((size_t)NPTS * HID * 4);
  __bf16* hb   = (__bf16*)take((size_t)NPTS * HID * 2);
  __bf16* sm1  = (__bf16*)take((size_t)3 * 256 * HID * 2);
  __bf16* sm2  = (__bf16*)take((size_t)3 * 128 * HID * 2);
  __bf16* su1  = (__bf16*)take((size_t)3 * 256 * HID * 2);
  __bf16* su2  = (__bf16*)take((size_t)3 * 128 * HID * 2);
  float*  mean = (float*)take(HID * 4);
  float*  rstd = (float*)take(HID * 4);
  float*  sums = (float*)take(2 * 4);

  for (int l = 0; l < 3; ++l) {
    swizzle_B<<<8, 256, 0, stream>>>(m1W + (size_t)l * 256 * HID, sm1 + (size_t)l * 256 * HID, 256);
    swizzle_B<<<4, 256, 0, stream>>>(m2W + (size_t)l * 128 * HID, sm2 + (size_t)l * 128 * HID, 128);
    swizzle_B<<<8, 256, 0, stream>>>(u1W + (size_t)l * 256 * HID, su1 + (size_t)l * 256 * HID, 256);
    swizzle_B<<<4, 256, 0, stream>>>(u2W + (size_t)l * 128 * HID, su2 + (size_t)l * 128 * HID, 128);
  }
  encode_kernel<<<NPTS * HID / 256, 256, 0, stream>>>(P, encW, encB, h, hb);

  for (int l = 0; l < 3; ++l) {
    hipMemsetAsync(agg, 0, (size_t)NPTS * HID * 4, stream);
    int mblocks = (E + 15) / 16;
    msg_kernel<<<mblocks, 256, 0, stream>>>(hb, src, dst,
                                            sm1 + (size_t)l * 256 * HID, m1B + l * HID,
                                            sm2 + (size_t)l * 128 * HID, m2B + l * HID,
                                            agg, E);
    upd_kernel<<<NPTS / 16, 256, 0, stream>>>(hb, agg,
                                              su1 + (size_t)l * 256 * HID, u1B + l * HID,
                                              su2 + (size_t)l * 128 * HID, u2B + l * HID, u);
    norm_stats_kernel<<<HID, 256, 0, stream>>>(u, mean, rstd);
    norm_apply_kernel<<<NPTS * HID / 256, 256, 0, stream>>>(u, mean, rstd, h, hb);
  }

  float* X = (float*)d_out + 1;
  decode_kernel<<<NPTS / 256, 256, 0, stream>>>(h, decW, decB, X);
  hipMemsetAsync(sums, 0, 8, stream);
  disc_kernel<<<NPTS / 256, 256, 0, stream>>>(X, sums);
  finalize_kernel<<<1, 1, 0, stream>>>(sums, (float*)d_out);
}
